// MetrixSoftmax_41137196761280
// MI455X (gfx1250) — matrix operand order repaired and checked by reading, not compiled, not run on hardware
//
#include <hip/hip_runtime.h>
#include <hip/hip_fp16.h>

typedef __attribute__((ext_vector_type(16))) _Float16 v16h;
typedef __attribute__((ext_vector_type(8)))  _Float16 v8h;
typedef __attribute__((ext_vector_type(8)))  float    v8f;
typedef __attribute__((ext_vector_type(4)))  float    v4f;

#define B_ROWS   8192
#define IN_DIM   256
#define OUT_DIM  16384
#define KSTEPS   8      // 256 / 32
#define WG_ROWS  32     // rows per workgroup (2 WMMA M-subtiles)
#define NITER    128    // 16384 / (8 waves * 16 cols)

// ---------------------------------------------------------------------------
// Kernel 1: f32 -> f16 conversion + row squared-norms. One wave32 per row.
// ---------------------------------------------------------------------------
__global__ void prep_rows(const float* __restrict__ src,
                          _Float16* __restrict__ dst,
                          float* __restrict__ norms, int nrows) {
  const int wave = threadIdx.x >> 5;
  const int lane = threadIdx.x & 31;
  const int row  = blockIdx.x * 8 + wave;
  if (row >= nrows) return;

  const float* s = src + (size_t)row * IN_DIM + lane * 8;
  v4f a = *(const v4f*)(s);
  v4f b = *(const v4f*)(s + 4);

  float ss = a.x*a.x + a.y*a.y + a.z*a.z + a.w*a.w
           + b.x*b.x + b.y*b.y + b.z*b.z + b.w*b.w;

  v8h h;
  h[0]=(_Float16)a.x; h[1]=(_Float16)a.y; h[2]=(_Float16)a.z; h[3]=(_Float16)a.w;
  h[4]=(_Float16)b.x; h[5]=(_Float16)b.y; h[6]=(_Float16)b.z; h[7]=(_Float16)b.w;
  *(v8h*)(dst + (size_t)row * IN_DIM + lane * 8) = h;

  // wave32 butterfly reduction
  #pragma unroll
  for (int off = 16; off > 0; off >>= 1) ss += __shfl_xor(ss, off, 32);
  if (lane == 0) norms[row] = ss;
}

// ---------------------------------------------------------------------------
// Kernel 2: fused  logits = -2*sqrt(max(x2 + w2 - 2*x.w, 0))  + softmax.
// One WG = 32 rows x full OUT dim. WMMA f16 -> f32 accumulate.
// ---------------------------------------------------------------------------
__global__ __launch_bounds__(256, 1)
void dist_softmax(const _Float16* __restrict__ xh,
                  const _Float16* __restrict__ wh,
                  const float* __restrict__ x2,
                  const float* __restrict__ w2,
                  float* __restrict__ out) {
  __shared__ float lds_x2[WG_ROWS];
  __shared__ float lds_part[WG_ROWS][128][2];   // per-row (max, sumexp) partials
  __shared__ float lds_M[WG_ROWS];
  __shared__ float lds_Sinv[WG_ROWS];

  const int tid   = threadIdx.x;
  const int wave  = tid >> 5;
  const int lane  = tid & 31;
  const int lhalf = lane >> 4;     // 0: lanes 0-15, 1: lanes 16-31
  const int l16   = lane & 15;
  const size_t rowBase = (size_t)blockIdx.x * WG_ROWS;

  if (tid < WG_ROWS) lds_x2[tid] = x2[rowBase + tid];
  __syncthreads();

  // ---- A matrix resident in VGPRs: 2 subtiles x 8 k-steps of 16x32 f16.
  // Layout used here (16-bit A 16x32, same k order as B): per k-step, each lane holds one contiguous
  // 16-half K block selected by lane half.
  v16h A[2][KSTEPS];
  #pragma unroll
  for (int t = 0; t < 2; ++t) {
    #pragma unroll
    for (int s = 0; s < KSTEPS; ++s) {
      const _Float16* base = xh + (rowBase + t*16 + l16) * IN_DIM + s*32 + lhalf*16;
      v8h lo = *(const v8h*)(base);
      v8h hi = *(const v8h*)(base + 8);
      v16h av;
      #pragma unroll
      for (int i = 0; i < 8; ++i) { av[i] = lo[i]; av[8+i] = hi[i]; }
      A[t][s] = av;
    }
  }

  // x2 per accumulator slot (row = t*16 + r + 8*lhalf), fixed for whole kernel
  float x2r[2][8];
  #pragma unroll
  for (int t = 0; t < 2; ++t)
    #pragma unroll
    for (int r = 0; r < 8; ++r)
      x2r[t][r] = lds_x2[t*16 + r + 8*lhalf];

  // online softmax state per slot
  float mrun[2][8], srun[2][8];
  #pragma unroll
  for (int t = 0; t < 2; ++t)
    #pragma unroll
    for (int r = 0; r < 8; ++r) { mrun[t][r] = -3.0e38f; srun[t][r] = 0.0f; }

  // ---------------- Phase 1: WMMA GEMM + logits + online (max,sum) ----------
  for (int it = 0; it < NITER; ++it) {
    const int colBase = it * 128 + wave * 16;

    // B (32x16 f16): lane half selects K 16-block, l16 selects column n;
    // column n is row n of w -> one contiguous 16-half chunk per lane.
    v16h Bv[KSTEPS];
    #pragma unroll
    for (int s = 0; s < KSTEPS; ++s)
      Bv[s] = *(const v16h*)(wh + (size_t)(colBase + l16) * IN_DIM + s*32 + lhalf*16);

    if (it + 1 < NITER)
      __builtin_prefetch(wh + (size_t)(colBase + 128 + l16) * IN_DIM, 0, 1);

    const float w2c = w2[colBase + l16];

    v8f c0 = {}; v8f c1 = {};
    #pragma unroll
    for (int s = 0; s < KSTEPS; ++s) {
      c0 = __builtin_amdgcn_wmma_f32_16x16x32_f16(false, A[0][s], false, Bv[s],
                                                  (short)0, c0, false, false);
      c1 = __builtin_amdgcn_wmma_f32_16x16x32_f16(false, A[1][s], false, Bv[s],
                                                  (short)0, c1, false, false);
    }

    // epilogue: distance -> logit -> online softmax -> store raw logit.
    // __builtin_amdgcn_sqrtf: raw v_sqrt_f32 (arg is clamped >= 0 and well
    // inside normal range, so the IEEE denorm-corrected expansion is pure
    // overhead).
    #pragma unroll
    for (int t = 0; t < 2; ++t) {
      #pragma unroll
      for (int r = 0; r < 8; ++r) {
        float dot = (t == 0) ? c0[r] : c1[r];
        float d2  = fmaxf(x2r[t][r] + w2c - 2.0f * dot, 0.0f);
        float logit = -2.0f * __builtin_amdgcn_sqrtf(d2);   // temperature = 2
        float mo = mrun[t][r];
        if (logit > mo) {
          srun[t][r] = srun[t][r] * __expf(mo - logit) + 1.0f;
          mrun[t][r] = logit;
        } else {
          srun[t][r] += __expf(logit - mo);
        }
        out[(rowBase + t*16 + r + 8*lhalf) * (size_t)OUT_DIM + colBase + l16] = logit;
      }
    }
  }

  // ---------------- Per-row reduction of 128 partials through LDS -----------
  #pragma unroll
  for (int t = 0; t < 2; ++t)
    #pragma unroll
    for (int r = 0; r < 8; ++r) {
      const int row = t*16 + r + 8*lhalf;
      lds_part[row][wave*16 + l16][0] = mrun[t][r];
      lds_part[row][wave*16 + l16][1] = srun[t][r];
    }
  __threadfence_block();   // make phase-1 global stores visible WG-wide
  __syncthreads();

  if (tid < WG_ROWS) {
    float M = -3.0e38f, S = 0.0f;
    for (int i = 0; i < 128; ++i) {
      float mi = lds_part[tid][i][0];
      float si = lds_part[tid][i][1];
      float Mn = fmaxf(M, mi);
      S = S * __expf(M - Mn) + si * __expf(mi - Mn);
      M = Mn;
    }
    lds_M[tid]    = M;
    lds_Sinv[tid] = 1.0f / S;
  }
  __syncthreads();

  // ---------------- Phase 2: renormalize own stripe (L2-hot re-read) --------
  float* stripe = out + rowBase * (size_t)OUT_DIM;
  const int nvec4 = WG_ROWS * OUT_DIM / 4;        // 131072 float4
  for (int i = tid; i < nvec4; i += 256) {
    v4f v = ((v4f*)stripe)[i];
    const int row = i >> 12;                      // (i*4) / 16384
    const float M = lds_M[row], Si = lds_Sinv[row];
    v.x = __expf(v.x - M) * Si;
    v.y = __expf(v.y - M) * Si;
    v.z = __expf(v.z - M) * Si;
    v.w = __expf(v.w - M) * Si;
    ((v4f*)stripe)[i] = v;
  }
}

// ---------------------------------------------------------------------------
extern "C" void kernel_launch(void* const* d_in, const int* in_sizes, int n_in,
                              void* d_out, int out_size, void* d_ws, size_t ws_size,
                              hipStream_t stream) {
  const float* x = (const float*)d_in[0];   // [8192, 256]
  const float* w = (const float*)d_in[1];   // [16384, 256]
  float* out = (float*)d_out;               // [8192, 16384]

  char* ws = (char*)d_ws;
  _Float16* xh = (_Float16*)ws;                                   // 4 MB
  _Float16* wh = (_Float16*)(ws + (size_t)B_ROWS * IN_DIM * 2);   // 8 MB
  float* x2 = (float*)(ws + (size_t)(B_ROWS + OUT_DIM) * IN_DIM * 2);
  float* w2 = x2 + B_ROWS;

  prep_rows<<<B_ROWS / 8, 256, 0, stream>>>(x, xh, x2, B_ROWS);
  prep_rows<<<OUT_DIM / 8, 256, 0, stream>>>(w, wh, w2, OUT_DIM);
  dist_softmax<<<B_ROWS / WG_ROWS, 256, 0, stream>>>(xh, wh, x2, w2, out);
}
